// decoder_cross_block_10161892623139
// MI455X (gfx1250) — compile-verified, matrix operand order checked by reading, not run on hardware
//
#include <hip/hip_runtime.h>

// ---------------------------------------------------------------------------
// MI455X (gfx1250) fused cross-attention block.
// Compute-bound on 3 GEMM stages (~480 GFLOP) -> v_wmma_f32_16x16x32_bf16.
// Global->LDS staging uses CDNA5 async copies (GLOBAL_LOAD_ASYNC_TO_LDS_B128,
// ASYNCcnt) so HBM latency overlaps WMMA work on the other LDS buffer.
// All B operands are pre-transposed to [N,K] so staging is pure 16B copies.
// ---------------------------------------------------------------------------

typedef __bf16 bf16;
typedef __attribute__((ext_vector_type(16))) __bf16 v16bf;
typedef __attribute__((ext_vector_type(8)))  __bf16 v8bf;
typedef __attribute__((ext_vector_type(4)))  __bf16 v4bf;
typedef __attribute__((ext_vector_type(8)))  float  v8f;
typedef __attribute__((ext_vector_type(4)))  int    v4i;

#define LDS_STRIDE 40   // 32 K-elems + 8 skew (80B rows: 16B aligned, bank-spread)

#if defined(__has_builtin)
#if __has_builtin(__builtin_amdgcn_global_load_async_to_lds_b128)
#define USE_ASYNC_COPY 1
#endif
#if __has_builtin(__builtin_amdgcn_s_wait_asynccnt)
#define HAVE_WAIT_ASYNC_BUILTIN 1
#endif
#endif

__device__ __forceinline__ void async_copy_b128(const bf16* g, bf16* l) {
#if USE_ASYNC_COPY
    // Signature (from clang diagnostic): (int4 addrspace(1)*, int4 addrspace(3)*, int, int)
    __builtin_amdgcn_global_load_async_to_lds_b128(
        (__attribute__((address_space(1))) v4i*)(bf16*)g,
        (__attribute__((address_space(3))) v4i*)l, /*offset=*/0, /*cpol=*/0);
#else
    *(v8bf*)l = *(const v8bf*)g;
#endif
}

__device__ __forceinline__ void wait_async_all() {
#if USE_ASYNC_COPY
#if HAVE_WAIT_ASYNC_BUILTIN
    __builtin_amdgcn_s_wait_asynccnt(0);
#else
    asm volatile("s_wait_asynccnt 0x0" ::: "memory");
#endif
#endif
}

__device__ __forceinline__ bf16 to_bf16(float f) {
    // round-to-nearest-even, bit-level
    unsigned u = __builtin_bit_cast(unsigned, f);
    unsigned r = (u + 0x7FFFu + ((u >> 16) & 1u)) >> 16;
    unsigned short h = (unsigned short)r;
    return __builtin_bit_cast(bf16, h);
}

// ---------------------------------------------------------------------------
// f32 -> bf16 streaming convert (grid-stride float4). n4 = n/4.
// ---------------------------------------------------------------------------
__global__ __launch_bounds__(256) void cvt_f32_bf16(const float* __restrict__ in,
                                                    bf16* __restrict__ out,
                                                    long long n4) {
    long long i = (long long)blockIdx.x * blockDim.x + threadIdx.x;
    long long stride = (long long)gridDim.x * blockDim.x;
    for (; i < n4; i += stride) {
        float4 v = ((const float4*)in)[i];
        v4bf o;
        o[0] = to_bf16(v.x); o[1] = to_bf16(v.y);
        o[2] = to_bf16(v.z); o[3] = to_bf16(v.w);
        ((v4bf*)out)[i] = o;
    }
}

// ---------------------------------------------------------------------------
// Fused convert + transpose: in f32 [R,C] -> out bf16 [C,R]. 32x32 LDS tile.
// R, C multiples of 32. grid = (C/32, R/32).
// ---------------------------------------------------------------------------
__global__ __launch_bounds__(256)
void transpose_cvt_f32_bf16(const float* __restrict__ in, bf16* __restrict__ out,
                            int R, int C) {
    __shared__ bf16 t[32][33];
    const int tx = threadIdx.x & 31, ty = threadIdx.x >> 5;  // ty 0..7
    const int c0 = blockIdx.x * 32, r0 = blockIdx.y * 32;
#pragma unroll
    for (int i = 0; i < 4; ++i) {
        int r = ty + i * 8;
        t[r][tx] = to_bf16(in[(long long)(r0 + r) * C + c0 + tx]);
    }
    __syncthreads();
#pragma unroll
    for (int i = 0; i < 4; ++i) {
        int c = ty + i * 8;
        out[(long long)(c0 + c) * R + r0 + tx] = t[tx][c];
    }
}

// ---------------------------------------------------------------------------
// Batched bf16 transpose: in [R,C] -> out [C,R] per batch z.
// grid = (C/32, R/32, batch).
// ---------------------------------------------------------------------------
__global__ __launch_bounds__(256)
void transpose_bf16(const bf16* __restrict__ in, bf16* __restrict__ out,
                    int R, int C, long long sIn, long long sOut) {
    __shared__ bf16 t[32][33];
    const bf16* ib = in + (long long)blockIdx.z * sIn;
    bf16* ob = out + (long long)blockIdx.z * sOut;
    const int tx = threadIdx.x & 31, ty = threadIdx.x >> 5;
    const int c0 = blockIdx.x * 32, r0 = blockIdx.y * 32;
#pragma unroll
    for (int i = 0; i < 4; ++i) {
        int r = ty + i * 8;
        t[r][tx] = ib[(long long)(r0 + r) * C + c0 + tx];
    }
    __syncthreads();
#pragma unroll
    for (int i = 0; i < 4; ++i) {
        int c = ty + i * 8;
        ob[(long long)(c0 + c) * R + r0 + tx] = t[tx][c];
    }
}

// ---------------------------------------------------------------------------
// bf16 WMMA GEMM: C[M,N](f32) = A[M,K] * B[N,K]^T  (both row-major, ld == K)
// EPI: 0 = +bias[n], 1 = *scale, 2 = none
// 256 thr (8 waves). Block tile 128x128, wave tile 64x32 (4x2 WMMA frags).
// Double-buffered LDS filled with async global->LDS b128 copies.
// ---------------------------------------------------------------------------
template <int EPI>
__global__ __launch_bounds__(256)
void gemm_bf16_wmma(const bf16* __restrict__ A, long long strideA,
                    const bf16* __restrict__ B, long long strideB,
                    float* __restrict__ C, long long strideC,
                    const float* __restrict__ bias,
                    int M, int N, int K, float scale) {
    __shared__ bf16 sA[2][128 * LDS_STRIDE];
    __shared__ bf16 sB[2][128 * LDS_STRIDE];

    const int tid = threadIdx.x;
    const int n0  = blockIdx.x * 128;
    const int m0  = blockIdx.y * 128;
    const int bz  = blockIdx.z;

    const bf16* Ab = A + (long long)bz * strideA;
    const bf16* Bb = B + (long long)bz * strideB;
    float*      Cb = C + (long long)bz * strideC;

    const int wave = tid >> 5;
    const int lane = tid & 31;
    const int half = lane >> 4;       // K-half selector (ISA frag layout)
    const int lr   = lane & 15;       // row (A) / col (B) within 16

    const int wm = (wave & 1) * 64;   // wave M offset in block tile
    const int wn = (wave >> 1) * 32;  // wave N offset in block tile

    v8f acc[4][2];
#pragma unroll
    for (int i = 0; i < 4; ++i)
#pragma unroll
        for (int j = 0; j < 2; ++j) acc[i][j] = (v8f)(0.0f);

    // Stage one 128x32 bf16 tile (A and B identical layout): 512 16B slots
    // each, 2 slots/thread/operand, all via async global->LDS copies.
    auto stage = [&](int buf, int k0) {
#pragma unroll
        for (int s = tid; s < 512; s += 256) {
            int r = s >> 2, c = (s & 3) * 8;
            async_copy_b128(&Ab[(long long)(m0 + r) * K + k0 + c],
                            &sA[buf][r * LDS_STRIDE + c]);
            async_copy_b128(&Bb[(long long)(n0 + r) * K + k0 + c],
                            &sB[buf][r * LDS_STRIDE + c]);
        }
    };

    const int nk = K >> 5;
    stage(0, 0);
    wait_async_all();
    __syncthreads();

    for (int t = 0; t < nk; ++t) {
        const int cur = t & 1, nxt = cur ^ 1;
        if (t + 1 < nk) stage(nxt, (t + 1) * 32);
        if (t + 2 < nk) { // L2-side prefetch of the tile after next
            __builtin_prefetch(&Ab[(long long)(m0 + (tid >> 1)) * K + (t + 2) * 32], 0, 1);
            __builtin_prefetch(&Bb[(long long)(n0 + (tid >> 1)) * K + (t + 2) * 32], 0, 1);
        }

        v16bf af[4], bfr[2];
#pragma unroll
        for (int i = 0; i < 4; ++i) {
            const int row = wm + i * 16 + lr;
            v8bf lo = *(const v8bf*)&sA[cur][row * LDS_STRIDE + half * 8];
            v8bf hi = *(const v8bf*)&sA[cur][row * LDS_STRIDE + 16 + half * 8];
            af[i] = __builtin_shufflevector(lo, hi, 0, 1, 2, 3, 4, 5, 6, 7,
                                            8, 9, 10, 11, 12, 13, 14, 15);
        }
#pragma unroll
        for (int j = 0; j < 2; ++j) {
            const int col = wn + j * 16 + lr;
            v8bf lo = *(const v8bf*)&sB[cur][col * LDS_STRIDE + half * 8];
            v8bf hi = *(const v8bf*)&sB[cur][col * LDS_STRIDE + 16 + half * 8];
            bfr[j] = __builtin_shufflevector(lo, hi, 0, 1, 2, 3, 4, 5, 6, 7,
                                             8, 9, 10, 11, 12, 13, 14, 15);
        }
#pragma unroll
        for (int i = 0; i < 4; ++i)
#pragma unroll
            for (int j = 0; j < 2; ++j)
                acc[i][j] = __builtin_amdgcn_wmma_f32_16x16x32_bf16(
                    false, af[i], false, bfr[j], (short)0, acc[i][j], false, false);

        wait_async_all();   // next buffer fully landed in LDS
        __syncthreads();
    }

    // C/D frag layout: VGPR v -> rows (v, v+8) across lane halves, col = lr
#pragma unroll
    for (int i = 0; i < 4; ++i) {
#pragma unroll
        for (int j = 0; j < 2; ++j) {
            const int col = n0 + wn + j * 16 + lr;
            float badd = 0.0f;
            if (EPI == 0) badd = bias[col];
#pragma unroll
            for (int v = 0; v < 8; ++v) {
                const int row = m0 + wm + i * 16 + half * 8 + v;
                float val = acc[i][j][v];
                if (EPI == 0) val += badd;
                if (EPI == 1) val *= scale;
                Cb[(long long)row * N + col] = val;
            }
        }
    }
}

// ---------------------------------------------------------------------------
// Row l2-normalize, H=1024, one 256-thread block per row (4 f32 / thread).
// ---------------------------------------------------------------------------
__global__ __launch_bounds__(256)
void rownorm_1024(float* __restrict__ inout, bf16* __restrict__ out_bf,
                  int write_f32) {
    __shared__ float red[256];
    const int tid = threadIdx.x;
    float* r = inout + (long long)blockIdx.x * 1024;
    float4 v = ((const float4*)r)[tid];
    red[tid] = v.x * v.x + v.y * v.y + v.z * v.z + v.w * v.w;
    __syncthreads();
#pragma unroll
    for (int off = 128; off > 0; off >>= 1) {
        if (tid < off) red[tid] += red[tid + off];
        __syncthreads();
    }
    const float rs = rsqrtf(fmaxf(red[0], 1e-12f));
    float4 o = make_float4(v.x * rs, v.y * rs, v.z * rs, v.w * rs);
    if (write_f32) ((float4*)r)[tid] = o;
    v4bf ob;
    ob[0] = to_bf16(o.x); ob[1] = to_bf16(o.y);
    ob[2] = to_bf16(o.z); ob[3] = to_bf16(o.w);
    ((v4bf*)(out_bf + (long long)blockIdx.x * 1024))[tid] = ob;
}

// ---------------------------------------------------------------------------
// Softmax over Lk=2048 in-place (f32) + bf16 copy. One block per row.
// ---------------------------------------------------------------------------
__global__ __launch_bounds__(256)
void softmax_2048(float* __restrict__ scores, bf16* __restrict__ probs_bf) {
    __shared__ float red[256];
    const int tid = threadIdx.x;
    float* r = scores + (long long)blockIdx.x * 2048;
    float4 a = ((const float4*)r)[tid];
    float4 b = ((const float4*)r)[tid + 256];
    float m = fmaxf(fmaxf(fmaxf(a.x, a.y), fmaxf(a.z, a.w)),
                    fmaxf(fmaxf(b.x, b.y), fmaxf(b.z, b.w)));
    red[tid] = m;
    __syncthreads();
#pragma unroll
    for (int off = 128; off > 0; off >>= 1) {
        if (tid < off) red[tid] = fmaxf(red[tid], red[tid + off]);
        __syncthreads();
    }
    const float rowmax = red[0];
    __syncthreads();
    a.x = __expf(a.x - rowmax); a.y = __expf(a.y - rowmax);
    a.z = __expf(a.z - rowmax); a.w = __expf(a.w - rowmax);
    b.x = __expf(b.x - rowmax); b.y = __expf(b.y - rowmax);
    b.z = __expf(b.z - rowmax); b.w = __expf(b.w - rowmax);
    red[tid] = a.x + a.y + a.z + a.w + b.x + b.y + b.z + b.w;
    __syncthreads();
#pragma unroll
    for (int off = 128; off > 0; off >>= 1) {
        if (tid < off) red[tid] += red[tid + off];
        __syncthreads();
    }
    const float inv = 1.0f / red[0];
    a.x *= inv; a.y *= inv; a.z *= inv; a.w *= inv;
    b.x *= inv; b.y *= inv; b.z *= inv; b.w *= inv;
    ((float4*)r)[tid] = a;
    ((float4*)r)[tid + 256] = b;
    bf16* rb = probs_bf + (long long)blockIdx.x * 2048;
    v4bf oa, ob;
    oa[0] = to_bf16(a.x); oa[1] = to_bf16(a.y); oa[2] = to_bf16(a.z); oa[3] = to_bf16(a.w);
    ob[0] = to_bf16(b.x); ob[1] = to_bf16(b.y); ob[2] = to_bf16(b.z); ob[3] = to_bf16(b.w);
    ((v4bf*)rb)[tid] = oa;
    ((v4bf*)rb)[tid + 256] = ob;
}

// ---------------------------------------------------------------------------
// cross = l2norm( qn + l2norm(emb_raw) ), H=1024, one block per row.
// ---------------------------------------------------------------------------
__global__ __launch_bounds__(256)
void cross_epilogue(const float* __restrict__ emb_raw,
                    const float* __restrict__ qn,
                    float* __restrict__ cross) {
    __shared__ float red[256];
    const int tid = threadIdx.x;
    const long long row = (long long)blockIdx.x * 1024;
    float4 e = ((const float4*)(emb_raw + row))[tid];
    red[tid] = e.x * e.x + e.y * e.y + e.z * e.z + e.w * e.w;
    __syncthreads();
#pragma unroll
    for (int off = 128; off > 0; off >>= 1) {
        if (tid < off) red[tid] += red[tid + off];
        __syncthreads();
    }
    const float rs1 = rsqrtf(fmaxf(red[0], 1e-12f));
    __syncthreads();
    float4 q = ((const float4*)(qn + row))[tid];
    float4 t = make_float4(q.x + e.x * rs1, q.y + e.y * rs1,
                           q.z + e.z * rs1, q.w + e.w * rs1);
    red[tid] = t.x * t.x + t.y * t.y + t.z * t.z + t.w * t.w;
    __syncthreads();
#pragma unroll
    for (int off = 128; off > 0; off >>= 1) {
        if (tid < off) red[tid] += red[tid + off];
        __syncthreads();
    }
    const float rs2 = rsqrtf(fmaxf(red[0], 1e-12f));
    float4 o = make_float4(t.x * rs2, t.y * rs2, t.z * rs2, t.w * rs2);
    ((float4*)(cross + row))[tid] = o;
}

// ---------------------------------------------------------------------------
extern "C" void kernel_launch(void* const* d_in, const int* in_sizes, int n_in,
                              void* d_out, int out_size, void* d_ws, size_t ws_size,
                              hipStream_t stream) {
    const float* Y  = (const float*)d_in[0];
    const float* X  = (const float*)d_in[1];
    const float* Wq = (const float*)d_in[2];
    const float* bq = (const float*)d_in[3];
    const float* Wk = (const float*)d_in[4];
    const float* bk = (const float*)d_in[5];
    const float* Wv = (const float*)d_in[6];
    const float* bv = (const float*)d_in[7];

    const int  Bn = 16, Lq = 2048, Lk = 2048, D = 1024, H = 1024;
    const long long MQ = (long long)Bn * Lq;  // 32768 query rows
    const long long MK = (long long)Bn * Lk;  // 32768 key rows

    // ---- workspace carve-out (256B aligned) -------------------------------
    char* ws = (char*)d_ws;
    size_t off = 0;
    auto carve = [&](size_t bytes) -> char* {
        char* p = ws + off;
        off += (bytes + 255) & ~(size_t)255;
        return p;
    };
    bf16*  Ybf    = (bf16*)carve(MQ * D * sizeof(bf16));
    bf16*  Xbf    = (bf16*)carve(MK * D * sizeof(bf16));
    bf16*  WqT    = (bf16*)carve((size_t)D * H * sizeof(bf16)); // [H,D]
    bf16*  WkT    = (bf16*)carve((size_t)D * H * sizeof(bf16));
    bf16*  WvT    = (bf16*)carve((size_t)D * H * sizeof(bf16));
    float* qraw   = (float*)carve(MQ * H * sizeof(float)); // -> qn f32 in-place
    float* kraw   = (float*)carve(MK * H * sizeof(float));
    float* vraw   = (float*)carve(MK * H * sizeof(float));
    bf16*  qn_bf  = (bf16*)carve(MQ * H * sizeof(bf16));
    bf16*  kn_bf  = (bf16*)carve(MK * H * sizeof(bf16));
    bf16*  vn_bf  = (bf16*)carve(MK * H * sizeof(bf16));
    bf16*  vnT    = (bf16*)carve(MK * H * sizeof(bf16));   // [Bn][H][Lk]
    bf16*  att_bf = (bf16*)carve((long long)Bn * Lq * Lk * sizeof(bf16));
    float* embraw = (float*)carve(MQ * H * sizeof(float));

    float* cross = (float*)d_out;          // [B,Lq,H]
    float* att   = cross + MQ * H;         // [B,Lq,Lk] (scores -> probs in place)

    // ---- 1. operand casts (+ weight transpose to [N,K]) -------------------
    cvt_f32_bf16<<<2048, 256, 0, stream>>>(Y, Ybf, MQ * D / 4);
    cvt_f32_bf16<<<2048, 256, 0, stream>>>(X, Xbf, MK * D / 4);
    transpose_cvt_f32_bf16<<<dim3(H / 32, D / 32), 256, 0, stream>>>(Wq, WqT, D, H);
    transpose_cvt_f32_bf16<<<dim3(H / 32, D / 32), 256, 0, stream>>>(Wk, WkT, D, H);
    transpose_cvt_f32_bf16<<<dim3(H / 32, D / 32), 256, 0, stream>>>(Wv, WvT, D, H);

    // ---- 2. projection GEMMs (bias epilogue), B = W^T [H,D] ---------------
    gemm_bf16_wmma<0><<<dim3(H / 128, (int)(MQ / 128), 1), 256, 0, stream>>>(
        Ybf, 0, WqT, 0, qraw, 0, bq, (int)MQ, H, D, 1.0f);
    gemm_bf16_wmma<0><<<dim3(H / 128, (int)(MK / 128), 1), 256, 0, stream>>>(
        Xbf, 0, WkT, 0, kraw, 0, bk, (int)MK, H, D, 1.0f);
    gemm_bf16_wmma<0><<<dim3(H / 128, (int)(MK / 128), 1), 256, 0, stream>>>(
        Xbf, 0, WvT, 0, vraw, 0, bv, (int)MK, H, D, 1.0f);

    // ---- 3. l2-normalize q,k,v (q also kept normalized in f32) ------------
    rownorm_1024<<<(int)MQ, 256, 0, stream>>>(qraw, qn_bf, 1);
    rownorm_1024<<<(int)MK, 256, 0, stream>>>(kraw, kn_bf, 0);
    rownorm_1024<<<(int)MK, 256, 0, stream>>>(vraw, vn_bf, 0);

    // ---- 4. transpose vn per batch: [Lk,H] -> [H,Lk] ----------------------
    transpose_bf16<<<dim3(H / 32, Lk / 32, Bn), 256, 0, stream>>>(
        vn_bf, vnT, Lk, H, (long long)Lk * H, (long long)H * Lk);

    // ---- 5. scores = qn * kn^T / sqrt(D) ----------------------------------
    gemm_bf16_wmma<1><<<dim3(Lk / 128, Lq / 128, Bn), 256, 0, stream>>>(
        qn_bf, (long long)Lq * H, kn_bf, (long long)Lk * H,
        att, (long long)Lq * Lk, nullptr, Lq, Lk, H, 0.03125f /*1/sqrt(1024)*/);

    // ---- 6. softmax (in-place f32 probs in d_out + bf16 copy) -------------
    softmax_2048<<<(int)MQ, 256, 0, stream>>>(att, att_bf);

    // ---- 7. emb_raw = att * vn  (B = vnT [H,Lk]) --------------------------
    gemm_bf16_wmma<2><<<dim3(H / 128, Lq / 128, Bn), 256, 0, stream>>>(
        att_bf, (long long)Lq * Lk, vnT, (long long)H * Lk,
        embraw, (long long)Lq * H, nullptr, Lq, H, Lk, 1.0f);

    // ---- 8. cross = l2norm(qn + l2norm(emb)) ------------------------------
    cross_epilogue<<<(int)MQ, 256, 0, stream>>>(embraw, qraw, cross);
}
